// Q2A_para_58299886076567
// MI455X (gfx1250) — matrix operand order repaired and checked by reading, not compiled, not run on hardware
//
#include <hip/hip_runtime.h>
#include <hip/hip_bf16.h>

// ---------------------------------------------------------------- types
typedef unsigned short u16;
typedef __attribute__((ext_vector_type(16))) __bf16 v16bf;
typedef __attribute__((ext_vector_type(8)))  float  v8f;
typedef __attribute__((ext_vector_type(8)))  unsigned short u16x8;

union BFrag { v16bf v; u16x8 h[2]; };

__device__ __forceinline__ u16 f2bf(float f) {
  unsigned u = __float_as_uint(f);
  if ((u & 0x7fffffffu) > 0x7f800000u) return (u16)0x7fc0;      // NaN
  unsigned r = (u + 0x7fffu + ((u >> 16) & 1u)) >> 16;          // RNE
  return (u16)r;
}
__device__ __forceinline__ float bf2f(u16 h) {
  return __uint_as_float(((unsigned)h) << 16);
}

// ---------------------------------------------------------------- GEMM
// C[M,N] = act(A[M,K](bf16,row-major) * B[N,K]^T(bf16,[N][K]) + bias[N])
// Wave grid 2x4; per-wave tile (16*MT) x (16*NT); block tile (32*MT) x (64*NT).
// Register double-buffered staging: raw global loads for tile k+1 issue before
// the WMMAs of tile k; bounds masking is deferred to the LDS-commit of the
// next iteration so the load-wait sits entirely behind matrix compute.
// Optional fused reduction (red_out != null): video segment-mean with softmax
// weights: red_out[b*768+col] += sum_rows(val) * red_w[b*16+seg] / 128
template<int MT, int NT>
__global__ __launch_bounds__(256)
void gemm_bf16_kernel(const u16* __restrict__ A, const u16* __restrict__ Bm,
                      const float* __restrict__ bias,
                      float* __restrict__ outF, u16* __restrict__ outB,
                      int M, int N, int K, int relu,
                      float* __restrict__ red_out, const float* __restrict__ red_w)
{
  constexpr int BM = 32 * MT;
  constexpr int BN = 64 * NT;
  constexpr int ACH = (BM * 4 + 255) / 256;   // per-thread A 16B chunks
  constexpr int BCH = (BN * 4 + 255) / 256;   // per-thread B 16B chunks
  __shared__ __align__(16) u16 As[BM * 32];
  __shared__ __align__(16) u16 Bs[BN * 32];

  const int tid  = threadIdx.x;
  const int wave = tid >> 5, lane = tid & 31;
  const int wm = wave >> 2, wn = wave & 3;           // 2 x 4 wave grid
  const int row0 = blockIdx.x * BM, col0 = blockIdx.y * BN;
  const int half = lane >> 4, l16 = lane & 15;
  const u16x8 zv = {0,0,0,0,0,0,0,0};

  // ---- per-thread staging descriptors (element offsets keep provenance) ----
  size_t aIdx[ACH]; int aOff[ACH]; bool aLive[ACH], aIn[ACH];
  #pragma unroll
  for (int i = 0; i < ACH; ++i) {
    const int idx = tid + 256 * i;
    aLive[i] = (idx < BM * 4);
    const int r = aLive[i] ? (idx >> 2) : 0;
    const int c = (idx & 3) * 8;
    const int g = row0 + r;
    aIn[i] = aLive[i] && (g < M);
    const int cg = g < M ? g : (M - 1);
    aIdx[i] = (size_t)cg * K + c;
    aOff[i] = r * 32 + c;
  }
  size_t bIdx[BCH]; int bOff[BCH]; bool bIn[BCH];
  #pragma unroll
  for (int i = 0; i < BCH; ++i) {
    const int idx = tid + 256 * i;
    const int r = idx >> 2, c = (idx & 3) * 8;
    const int g = col0 + r;
    bIn[i] = (g < N);
    const int cg = g < N ? g : (N - 1);
    bIdx[i] = (size_t)cg * K + c;
    bOff[i] = r * 32 + c;
  }

  v8f acc[MT][NT] = {};
  u16x8 aReg[ACH], bReg[BCH];
  #pragma unroll
  for (int i = 0; i < ACH; ++i) aReg[i] = *(const u16x8*)(A + aIdx[i]);
  #pragma unroll
  for (int i = 0; i < BCH; ++i) bReg[i] = *(const u16x8*)(Bm + bIdx[i]);

  for (int k0 = 0; k0 < K; k0 += 32) {
    // ---- commit staged registers to LDS (apply loop-invariant masks here) --
    #pragma unroll
    for (int i = 0; i < ACH; ++i)
      if (aLive[i]) *(u16x8*)(&As[aOff[i]]) = aIn[i] ? aReg[i] : zv;
    #pragma unroll
    for (int i = 0; i < BCH; ++i)
      *(u16x8*)(&Bs[bOff[i]]) = bIn[i] ? bReg[i] : zv;
    __syncthreads();

    // ---- issue next-tile raw global loads (no wait until next commit) ----
    if (k0 + 32 < K) {
      #pragma unroll
      for (int i = 0; i < ACH; ++i) {
        aIdx[i] += 32;
        aReg[i] = *(const u16x8*)(A + aIdx[i]);
        __builtin_prefetch(A + aIdx[i] + 32, 0, 3);
      }
      #pragma unroll
      for (int i = 0; i < BCH; ++i) {
        bIdx[i] += 32;
        bReg[i] = *(const u16x8*)(Bm + bIdx[i]);
        __builtin_prefetch(Bm + bIdx[i] + 32, 0, 3);
      }
    }

    // ---- fragments (ISA 7.12.2 layouts) + WMMA ----
    BFrag af[MT], bfr[NT];
    #pragma unroll
    for (int mt = 0; mt < MT; ++mt) {
      const int arow = wm * 16 * MT + mt * 16 + l16;
      af[mt].h[0] = *(const u16x8*)(&As[arow * 32 + 8 * half]);      // K=8h..
      af[mt].h[1] = *(const u16x8*)(&As[arow * 32 + 16 + 8 * half]); // K=16+8h..
    }
    #pragma unroll
    for (int nt = 0; nt < NT; ++nt) {
      const int bcol = wn * 16 * NT + nt * 16 + l16;
      bfr[nt].h[0] = *(const u16x8*)(&Bs[bcol * 32 + 8 * half]);      // K=8h..
      bfr[nt].h[1] = *(const u16x8*)(&Bs[bcol * 32 + 16 + 8 * half]);
    }
    #pragma unroll
    for (int mt = 0; mt < MT; ++mt)
      #pragma unroll
      for (int nt = 0; nt < NT; ++nt)
        acc[mt][nt] = __builtin_amdgcn_wmma_f32_16x16x32_bf16(
            false, af[mt].v, false, bfr[nt].v, (short)0, acc[mt][nt], false, false);
    __syncthreads();
  }

  if (red_out) {   // fused video segment reduction (M=16384, N=768 only)
    #pragma unroll
    for (int mt = 0; mt < MT; ++mt) {
      const int rbase = row0 + wm * 16 * MT + mt * 16 + 8 * half;
      const int b   = rbase >> 11;           // row / 2048
      const int seg = (rbase & 2047) >> 7;   // (row % 2048) / 128
      const float w = red_w[b * 16 + seg] * (1.0f / 128.0f);
      #pragma unroll
      for (int nt = 0; nt < NT; ++nt) {
        const int col = col0 + wn * 16 * NT + nt * 16 + l16;
        if (col >= N) continue;
        const float bv = bias ? bias[col] : 0.f;
        float s = 0.f;
        #pragma unroll
        for (int r = 0; r < 8; ++r) s += acc[mt][nt][r] + bv;
        atomicAdd(&red_out[b * 768 + col], s * w);
      }
    }
    return;
  }

  #pragma unroll
  for (int mt = 0; mt < MT; ++mt) {
    const int rbase = row0 + wm * 16 * MT + mt * 16 + 8 * half;
    #pragma unroll
    for (int nt = 0; nt < NT; ++nt) {
      const int col = col0 + wn * 16 * NT + nt * 16 + l16;
      if (col >= N) continue;
      const float bv = bias ? bias[col] : 0.f;
      #pragma unroll
      for (int r = 0; r < 8; ++r) {
        const int row = rbase + r;
        if (row >= M) continue;
        float v = acc[mt][nt][r] + bv;
        if (relu) v = fmaxf(v, 0.f);
        if (outF) outF[(size_t)row * N + col] = v;
        if (outB) outB[(size_t)row * N + col] = f2bf(v);
      }
    }
  }
}

// ---------------------------------------------------------------- helpers
// transpose + convert: out[n*K+k] = bf16(W[k*N+n])   (W is [K,N] fp32)
__global__ void convT_kernel(const float* __restrict__ W, u16* __restrict__ out,
                             int K, int N) {
  const size_t total = (size_t)K * N;
  for (size_t i = (size_t)blockIdx.x * blockDim.x + threadIdx.x; i < total;
       i += (size_t)gridDim.x * blockDim.x) {
    const size_t n = i / K, k = i % K;
    out[i] = f2bf(W[k * (size_t)N + n]);
  }
}
// plain convert
__global__ void convN_kernel(const float* __restrict__ in, u16* __restrict__ out,
                             size_t n) {
  for (size_t i = (size_t)blockIdx.x * blockDim.x + threadIdx.x; i < n;
       i += (size_t)gridDim.x * blockDim.x)
    out[i] = f2bf(in[i]);
}

__global__ void zero_kernel(float* p, int n) {
  int i = blockIdx.x * blockDim.x + threadIdx.x;
  if (i < n) p[i] = 0.f;
}

__global__ void softmax16_kernel(const float* __restrict__ x, float* __restrict__ y) {
  const int b = blockIdx.x;
  float m = x[b * 16];
  for (int i = 1; i < 16; ++i) m = fmaxf(m, x[b * 16 + i]);
  float e[16], s = 0.f;
  for (int i = 0; i < 16; ++i) { e[i] = __expf(x[b * 16 + i] - m); s += e[i]; }
  const float inv = 1.f / s;
  for (int i = 0; i < 16; ++i) y[b * 16 + i] = e[i] * inv;
}

// text_in[b, 0]=question, text_in[b, 1+p]=para[p]   (bf16 rows of 768)
__global__ void pack_text_kernel(const float* __restrict__ q,
                                 const float* __restrict__ par,
                                 u16* __restrict__ out) {
  const int row = blockIdx.x, d = threadIdx.x;   // grid = B*17, block = 768
  const int b = row / 17, p = row % 17;
  const float v = (p == 0) ? q[b * 768 + d] : par[((size_t)b * 16 + (p - 1)) * 768 + d];
  out[(size_t)row * 768 + d] = f2bf(v);
}

// pr[b,d] = sum_p ps[b,p] * text_out[b, 1+p, d]
__global__ void pr_kernel(const float* __restrict__ ps, const float* __restrict__ text_out,
                          float* __restrict__ pr) {
  const int b = blockIdx.x, d = threadIdx.x;     // grid = B, block = 768
  float s = 0.f;
  for (int p = 0; p < 16; ++p)
    s += ps[b * 16 + p] * text_out[((size_t)b * 17 + 1 + p) * 768 + d];
  pr[b * 768 + d] = s;
}

__global__ void hinit_kernel(const float* __restrict__ s0, u16* __restrict__ h) {
  const int b = blockIdx.x, d = threadIdx.x;     // grid = B, block = 768
  h[b * 768 + d] = f2bf(s0[d]);
}

// gather step-s answer rows [B,A,768] -> bf16, from [B,S,A,768]
__global__ void pack_step_kernel(const float* __restrict__ at, const float* __restrict__ ab,
                                 u16* __restrict__ at_bf, u16* __restrict__ ab_bf, int s) {
  const int row = blockIdx.x, d = threadIdx.x;   // grid = 128, block = 768
  const int b = row >> 4, a = row & 15;
  const size_t src = (((size_t)b * 8 + s) * 16 + a) * 768 + d;
  at_bf[(size_t)row * 768 + d] = f2bf(at[src]);
  ab_bf[(size_t)row * 768 + d] = f2bf(ab[src]);
}

// inp[row] = [ vp(b) | pr(b) | q(b)+a_t(row) | a_b(row) ]  -> bf16 [128,3072]
__global__ void pack_inp_kernel(const float* __restrict__ vp, const float* __restrict__ pr,
                                const float* __restrict__ text_out,
                                const float* __restrict__ a_t, const float* __restrict__ a_b,
                                u16* __restrict__ inp) {
  const int row = blockIdx.x, d = threadIdx.x;   // grid = 128, block = 768
  const int b = row >> 4;
  u16* o = inp + (size_t)row * 3072;
  o[d]        = f2bf(vp[b * 768 + d]);
  o[768 + d]  = f2bf(pr[b * 768 + d]);
  o[1536 + d] = f2bf(text_out[(size_t)b * 17 * 768 + d] + a_t[(size_t)row * 768 + d]);
  o[2304 + d] = f2bf(a_b[(size_t)row * 768 + d]);
}

// GRUCell gates (elementwise over 128*768)
__global__ void gru_gate_kernel(const float* __restrict__ gi, const float* __restrict__ gh,
                                const u16* __restrict__ h_bf, u16* __restrict__ states) {
  const int idx = blockIdx.x * blockDim.x + threadIdx.x;
  if (idx >= 128 * 768) return;
  const int row = idx / 768, d = idx % 768;
  const int b = row >> 4;
  const float ir = gi[(size_t)row * 2304 + d];
  const float iz = gi[(size_t)row * 2304 + 768 + d];
  const float in_ = gi[(size_t)row * 2304 + 1536 + d];
  const float hr = gh[(size_t)b * 2304 + d];
  const float hz = gh[(size_t)b * 2304 + 768 + d];
  const float hn = gh[(size_t)b * 2304 + 1536 + d];
  const float r = 1.f / (1.f + __expf(-(ir + hr)));
  const float z = 1.f / (1.f + __expf(-(iz + hz)));
  const float n = tanhf(in_ + r * hn);
  const float h = bf2f(h_bf[b * 768 + d]);
  states[idx] = f2bf((1.f - z) * n + z * h);
}

// logits[row] = proj_hid[row,:] . pr_w2 + pr_b2 ; also scatter into d_out[B,S,A]
__global__ void gemv_logits_kernel(const u16* __restrict__ hid, const float* __restrict__ w,
                                   const float* __restrict__ b2, float* __restrict__ scores,
                                   float* __restrict__ logits, int s) {
  const int row = blockIdx.x;        // 0..127
  const int lane = threadIdx.x;      // 32
  float sum = 0.f;
  for (int d = lane; d < 768; d += 32)
    sum += bf2f(hid[(size_t)row * 768 + d]) * w[d];
  for (int off = 16; off; off >>= 1) sum += __shfl_down(sum, off);
  if (lane == 0) {
    const float v = sum + b2[0];
    logits[row] = v;
    const int b = row >> 4, a = row & 15;
    scores[((size_t)b * 8 + s) * 16 + a] = v;
  }
}

// per batch: h <- x[argmax(logits)]
__global__ void argmax_select_kernel(const float* __restrict__ logits,
                                     const u16* __restrict__ x_bf, u16* __restrict__ h_bf) {
  const int b = blockIdx.x;
  __shared__ int amax;
  if (threadIdx.x == 0) {
    float best = logits[b * 16]; int bi = 0;
    for (int a = 1; a < 16; ++a) {
      const float v = logits[b * 16 + a];
      if (v > best) { best = v; bi = a; }
    }
    amax = bi;
  }
  __syncthreads();
  const size_t src = ((size_t)b * 16 + amax) * 768;
  for (int d = threadIdx.x; d < 768; d += blockDim.x)
    h_bf[b * 768 + d] = x_bf[src + d];
}

// ---------------------------------------------------------------- launch
static inline void gemm(hipStream_t st, bool big, const u16* A, const u16* Bm,
                        const float* bias, float* outF, u16* outB,
                        int M, int N, int K, int relu,
                        float* red_out = nullptr, const float* red_w = nullptr) {
  if (big) {      // 64 x 256 block tile, 8 wmma / wave / K-step
    dim3 g((M + 63) / 64, (N + 255) / 256);
    gemm_bf16_kernel<2, 4><<<g, 256, 0, st>>>(A, Bm, bias, outF, outB, M, N, K,
                                              relu, red_out, red_w);
  } else {        // 32 x 128 block tile, more blocks in flight
    dim3 g((M + 31) / 32, (N + 127) / 128);
    gemm_bf16_kernel<1, 2><<<g, 256, 0, st>>>(A, Bm, bias, outF, outB, M, N, K,
                                              relu, red_out, red_w);
  }
}

extern "C" void kernel_launch(void* const* d_in, const int* in_sizes, int n_in,
                              void* d_out, int out_size, void* d_ws, size_t ws_size,
                              hipStream_t stream) {
  constexpr int Bn = 8, T = 2048, P = 16, Sn = 8, An = 16, D = 768, H = 768;
  constexpr int D4 = 4 * D, H3 = 3 * H;

  const float* video    = (const float*)d_in[0];
  const float* para     = (const float*)d_in[1];
  const float* question = (const float*)d_in[2];
  const float* pscore   = (const float*)d_in[3];
  const float* a_texts  = (const float*)d_in[4];
  const float* a_btn    = (const float*)d_in[5];
  const float* mv_w1 = (const float*)d_in[6];  const float* mv_b1 = (const float*)d_in[7];
  const float* mv_w2 = (const float*)d_in[8];  const float* mv_b2 = (const float*)d_in[9];
  const float* mt_w1 = (const float*)d_in[10]; const float* mt_b1 = (const float*)d_in[11];
  const float* mt_w2 = (const float*)d_in[12]; const float* mt_b2 = (const float*)d_in[13];
  const float* mp_w1 = (const float*)d_in[14]; const float* mp_b1 = (const float*)d_in[15];
  const float* mp_w2 = (const float*)d_in[16]; const float* mp_b2 = (const float*)d_in[17];
  const float* g_wih = (const float*)d_in[18]; const float* g_bih = (const float*)d_in[19];
  const float* g_whh = (const float*)d_in[20]; const float* g_bhh = (const float*)d_in[21];
  const float* pr_w1 = (const float*)d_in[22]; const float* pr_b1 = (const float*)d_in[23];
  const float* pr_w2 = (const float*)d_in[24]; const float* pr_b2 = (const float*)d_in[25];
  const float* state0 = (const float*)d_in[26];
  float* scores = (float*)d_out;   // [B, S, A] fp32

  // ---- workspace carve-out ----
  char* base = (char*)d_ws;  size_t off = 0;
  auto alloc = [&](size_t bytes) -> void* {
    void* p = base + off;  off += (bytes + 255) & ~(size_t)255;  return p;
  };
  u16* w_mv1t = (u16*)alloc((size_t)D * D * 2);
  u16* w_mv2t = (u16*)alloc((size_t)D * D * 2);
  u16* w_mt1t = (u16*)alloc((size_t)D * D * 2);
  u16* w_mt2t = (u16*)alloc((size_t)D * D * 2);
  u16* w_pr1t = (u16*)alloc((size_t)H * H * 2);
  u16* w_mp1t = (u16*)alloc((size_t)D4 * D4 * 2);
  u16* w_mp2t = (u16*)alloc((size_t)H * D4 * 2);     // [N=H][K=D4]
  u16* w_wih  = (u16*)alloc((size_t)H3 * H * 2);     // already [N][K]
  u16* w_whh  = (u16*)alloc((size_t)H3 * H * 2);
  u16* video_bf = (u16*)alloc((size_t)Bn * T * D * 2);
  u16* hidA     = (u16*)alloc((size_t)Bn * T * D * 2);
  u16* text_in  = (u16*)alloc((size_t)Bn * 17 * D * 2);
  u16* text_tmp = (u16*)alloc((size_t)Bn * 17 * D * 2);
  float* text_out = (float*)alloc((size_t)Bn * 17 * D * 4);
  float* ps_soft = (float*)alloc(Bn * P * 4);
  float* vp = (float*)alloc((size_t)Bn * D * 4);
  float* pr = (float*)alloc((size_t)Bn * D * 4);
  u16* h_bf   = (u16*)alloc((size_t)Bn * H * 2);
  u16* at_bf  = (u16*)alloc((size_t)128 * D * 2);
  u16* ab_bf  = (u16*)alloc((size_t)128 * D * 2);
  u16* mtmp   = (u16*)alloc((size_t)128 * D * 2);
  float* a_t_out = (float*)alloc((size_t)128 * D * 4);
  float* a_b_out = (float*)alloc((size_t)128 * D * 4);
  u16* inp_bf  = (u16*)alloc((size_t)128 * D4 * 2);
  u16* pre_hid = (u16*)alloc((size_t)128 * D4 * 2);
  u16* x_bf    = (u16*)alloc((size_t)128 * H * 2);
  float* gi = (float*)alloc((size_t)128 * H3 * 4);
  float* gh = (float*)alloc((size_t)Bn * H3 * 4);
  u16* states_bf = (u16*)alloc((size_t)128 * H * 2);
  float* logits = (float*)alloc(128 * 4);
  (void)ws_size; (void)n_in; (void)in_sizes; (void)out_size;

  // ---- weight conversion (fp32 -> bf16, transposed to [N][K]) ----
  auto convT = [&](const float* W, u16* o, int K, int N) {
    size_t tot = (size_t)K * N;
    int grid = (int)((tot + 255) / 256); if (grid > 8192) grid = 8192;
    convT_kernel<<<grid, 256, 0, stream>>>(W, o, K, N);
  };
  convT(mv_w1, w_mv1t, D, D);   convT(mv_w2, w_mv2t, D, D);
  convT(mt_w1, w_mt1t, D, D);   convT(mt_w2, w_mt2t, D, D);
  convT(pr_w1, w_pr1t, H, H);
  convT(mp_w1, w_mp1t, D4, D4); convT(mp_w2, w_mp2t, D4, H);
  {
    size_t n = (size_t)H3 * H;
    convN_kernel<<<4096, 256, 0, stream>>>(g_wih, w_wih, n);
    convN_kernel<<<4096, 256, 0, stream>>>(g_whh, w_whh, n);
    convN_kernel<<<8192, 256, 0, stream>>>(video, video_bf, (size_t)Bn * T * D);
  }

  // ---- softmax(paras_score), vp via fused video MLP reduction ----
  softmax16_kernel<<<Bn, 1, 0, stream>>>(pscore, ps_soft);
  zero_kernel<<<(Bn * D + 255) / 256, 256, 0, stream>>>(vp, Bn * D);
  gemm(stream, true, video_bf, w_mv1t, mv_b1, nullptr, hidA, Bn * T, D, D, 1);
  gemm(stream, true, hidA, w_mv2t, mv_b2, nullptr, nullptr, Bn * T, D, D, 0, vp, ps_soft);

  // ---- text MLP on [question; para] rows, then pr ----
  pack_text_kernel<<<Bn * 17, D, 0, stream>>>(question, para, text_in);
  gemm(stream, false, text_in, w_mt1t, mt_b1, nullptr, text_tmp, Bn * 17, D, D, 1);
  gemm(stream, false, text_tmp, w_mt2t, mt_b2, text_out, nullptr, Bn * 17, D, D, 0);
  pr_kernel<<<Bn, D, 0, stream>>>(ps_soft, text_out, pr);

  // ---- initial GRU state ----
  hinit_kernel<<<Bn, H, 0, stream>>>(state0, h_bf);

  // ---- sequential scan over S steps ----
  for (int s = 0; s < Sn; ++s) {
    pack_step_kernel<<<Bn * An, D, 0, stream>>>(a_texts, a_btn, at_bf, ab_bf, s);
    // a_t = mlp_t(at), a_b = mlp_v(ab)
    gemm(stream, false, at_bf, w_mt1t, mt_b1, nullptr, mtmp, 128, D, D, 1);
    gemm(stream, false, mtmp, w_mt2t, mt_b2, a_t_out, nullptr, 128, D, D, 0);
    gemm(stream, false, ab_bf, w_mv1t, mv_b1, nullptr, mtmp, 128, D, D, 1);
    gemm(stream, false, mtmp, w_mv2t, mv_b2, a_b_out, nullptr, 128, D, D, 0);
    // inp = [vp | pr | q + a_t | a_b]
    pack_inp_kernel<<<Bn * An, D, 0, stream>>>(vp, pr, text_out, a_t_out, a_b_out, inp_bf);
    // x = mlp_pre(inp)
    gemm(stream, false, inp_bf, w_mp1t, mp_b1, nullptr, pre_hid, 128, D4, D4, 1);
    gemm(stream, false, pre_hid, w_mp2t, mp_b2, nullptr, x_bf, 128, H, D4, 0);
    // GRU gates
    gemm(stream, false, x_bf, w_wih, g_bih, gi, nullptr, 128, H3, H, 0);
    gemm(stream, false, h_bf, w_whh, g_bhh, gh, nullptr, Bn, H3, H, 0);
    gru_gate_kernel<<<(128 * H + 255) / 256, 256, 0, stream>>>(gi, gh, h_bf, states_bf);
    // logits = proj(states)
    gemm(stream, false, states_bf, w_pr1t, pr_b1, nullptr, mtmp, 128, H, H, 1);
    gemv_logits_kernel<<<128, 32, 0, stream>>>(mtmp, pr_w2, pr_b2, scores, logits, s);
    // h <- x[argmax]
    argmax_select_kernel<<<Bn, 256, 0, stream>>>(logits, x_bf, h_bf);
  }
}
